// Infer_CascadeREDNet_9758165696781
// MI455X (gfx1250) — matrix operand order repaired and checked by reading, not compiled, not run on hardware
//
#include <hip/hip_runtime.h>
#include <hip/hip_bf16.h>
#include <stdint.h>

typedef __attribute__((ext_vector_type(16))) _Float16 v16h;
typedef __attribute__((ext_vector_type(8)))  _Float16 v8h;
typedef __attribute__((ext_vector_type(8)))  float    v8f;

#define HH 320
#define WW 320
#define N1 (HH * WW)

static __device__ __forceinline__ int imax(int a, int b) { return a > b ? a : b; }
static __device__ __forceinline__ int imin(int a, int b) { return a < b ? a : b; }

// ---------------------------------------------------------------------------
// Zero-fill (u32 granularity)
// ---------------------------------------------------------------------------
__global__ void zero_kernel(uint32_t* __restrict__ p, int n) {
  int i = blockIdx.x * blockDim.x + threadIdx.x;
  if (i < n) p[i] = 0u;
}

// ---------------------------------------------------------------------------
// Projection prep: inv(P0), rel = P_v * inv(P0) -> rot(3x3), trans(3) per view
// ---------------------------------------------------------------------------
__global__ void prep_proj_kernel(const float* __restrict__ proj, float* __restrict__ rt) {
  if (threadIdx.x != 0 || blockIdx.x != 0) return;
  float a[4][8];
  for (int i = 0; i < 4; ++i)
    for (int j = 0; j < 4; ++j) {
      a[i][j] = proj[i * 4 + j];
      a[i][j + 4] = (i == j) ? 1.f : 0.f;
    }
  for (int c = 0; c < 4; ++c) {
    int piv = c;
    for (int r = c + 1; r < 4; ++r)
      if (fabsf(a[r][c]) > fabsf(a[piv][c])) piv = r;
    if (piv != c)
      for (int j = 0; j < 8; ++j) { float t = a[c][j]; a[c][j] = a[piv][j]; a[piv][j] = t; }
    float d = a[c][c];
    for (int j = 0; j < 8; ++j) a[c][j] /= d;
    for (int r = 0; r < 4; ++r) {
      if (r == c) continue;
      float f = a[r][c];
      for (int j = 0; j < 8; ++j) a[r][j] -= f * a[c][j];
    }
  }
  float inv[4][4];
  for (int i = 0; i < 4; ++i)
    for (int j = 0; j < 4; ++j) inv[i][j] = a[i][j + 4];
  for (int v = 0; v < 2; ++v) {
    const float* P = proj + (v + 1) * 16;
    float rel[4][4];
    for (int i = 0; i < 4; ++i)
      for (int j = 0; j < 4; ++j) {
        float s = 0.f;
        for (int k = 0; k < 4; ++k) s += P[i * 4 + k] * inv[k][j];
        rel[i][j] = s;
      }
    for (int i = 0; i < 3; ++i)
      for (int j = 0; j < 3; ++j) rt[v * 12 + i * 3 + j] = rel[i][j];
    for (int i = 0; i < 3; ++i) rt[v * 12 + 9 + i] = rel[i][3];
  }
}

// ---------------------------------------------------------------------------
// Pack weights into WMMA A-matrix layout, TAP-MAJOR K ordering:
//   x region:  k in [0, Kx)        : tap = k >> lgx, c = k & (Cinx-1)
//   pad:       k in [Kx, KxP)      : 0      (KxP = Kx rounded up to 32)
//   s region:  k in [KxP, KxP+Ks)  : tap = ks >> lgs, c = ks & (Cins-1)
// A element for lane L (m=L&15, hi=L>>4), half h:
//   K = ((h&8)<<1) + hi*8 + (h&7),  M = mtile*16 + m
// dst index = ((mtile*ksteps + kstep)*32 + lane)*16 + h
// ---------------------------------------------------------------------------
__global__ void pack_weights_kernel(const float* __restrict__ Wx, const float* __restrict__ Ws,
                                    int Cout, int Cinx, int lgx, int Cins, int lgs,
                                    int taps, int Kx, int KxP,
                                    int ksteps, int mtiles, _Float16* __restrict__ dst) {
  int t = blockIdx.x * blockDim.x + threadIdx.x;
  int total = mtiles * ksteps * 512;
  if (t >= total) return;
  int h    = t & 15;
  int lane = (t >> 4) & 31;
  int tile = t >> 9;
  int kstep = tile % ksteps;
  int mtile = tile / ksteps;
  int m  = mtile * 16 + (lane & 15);
  int hi = lane >> 4;
  int k  = kstep * 32 + ((h & 8) << 1) + hi * 8 + (h & 7);
  float v = 0.f;
  if (m < Cout) {
    if (k < Kx) {
      int c = k & (Cinx - 1), tap = k >> lgx;
      v = Wx[(m * Cinx + c) * taps + tap];
    } else if (k >= KxP && k < KxP + Cins * taps) {
      int ks = k - KxP;
      int c = ks & (Cins - 1), tap = ks >> lgs;
      v = Ws[(m * Cins + c) * taps + tap];
    }
  }
  dst[t] = (_Float16)v;
}

// ---------------------------------------------------------------------------
// Homography warp + variance cost volume (f16 out, [c][pixel])
// ---------------------------------------------------------------------------
__global__ void warp_variance_kernel(const float* __restrict__ feats,
                                     const float* __restrict__ rt,
                                     const float* __restrict__ dvals, int d,
                                     _Float16* __restrict__ var) {
  int p = blockIdx.x * blockDim.x + threadIdx.x;
  if (p >= N1) return;
  int oy = p / WW, ox = p - oy * WW;
  float depth = dvals[d * N1 + p];
  int   offs[2][4];
  float wts[2][4];
  for (int v = 0; v < 2; ++v) {
    const float* R = rt + v * 12;
    float X = (R[0] * ox + R[1] * oy + R[2]) * depth + R[9];
    float Y = (R[3] * ox + R[4] * oy + R[5]) * depth + R[10];
    float Z = (R[6] * ox + R[7] * oy + R[8]) * depth + R[11];
    float px = X / Z, py = Y / Z;
    float x0 = floorf(px), y0 = floorf(py);
    float fx = px - x0, fy = py - y0;
    for (int t = 0; t < 4; ++t) {
      int a = t & 1, b = t >> 1;
      float xt = x0 + (float)a, yt = y0 + (float)b;
      bool ok = (xt >= 0.f) && (xt <= (float)(WW - 1)) && (yt >= 0.f) && (yt <= (float)(HH - 1));
      int xc = imin(imax((int)xt, 0), WW - 1);
      int yc = imin(imax((int)yt, 0), HH - 1);
      offs[v][t] = yc * WW + xc;
      float w = (a ? fx : 1.f - fx) * (b ? fy : 1.f - fy);
      wts[v][t] = ok ? w : 0.f;
    }
  }
  for (int c = 0; c < 32; ++c) {
    float ref = feats[c * N1 + p];
    float sum = ref, sq = ref * ref;
    for (int v = 0; v < 2; ++v) {
      const float* img = feats + (size_t)((v + 1) * 32 + c) * N1;
      float w = wts[v][0] * img[offs[v][0]] + wts[v][1] * img[offs[v][1]] +
                wts[v][2] * img[offs[v][2]] + wts[v][3] * img[offs[v][3]];
      sum += w;
      sq  += w * w;
    }
    float m  = sum * (1.f / 3.f);
    float vr = sq * (1.f / 3.f) - m * m;
    var[(size_t)c * N1 + p] = (_Float16)vr;
  }
}

// ---------------------------------------------------------------------------
// LDS-staged fused-conv GEMM via V_WMMA_F32_16X16X32_F16.
// Block = 160 threads (5 waves). Each block computes one full-width band of
// 320 output pixels (BH = 320/OW rows). Wave w computes 4 N-tiles; all tiles
// valid by construction (EXEC all-1s at every WMMA).
// Inputs staged to dynamic LDS in channel-innermost [row][col][c] layout with
// pre-zeroed halos -> B fragment = 2x 16-byte ds loads per lane per k-step,
// no bounds checks in the hot loop. Tap-major K: tap = k>>lg, c = k&(CIN-1).
// ---------------------------------------------------------------------------
template <int TAPS, int STRIDE, int CINX, int CINS, int COUT,
          int OH, int OW, bool RELU, bool OUTH_EN, bool OUTF_EN>
__global__ __launch_bounds__(160) void conv_wmma_kernel(
    const _Float16* __restrict__ Apk, const _Float16* __restrict__ x,
    const _Float16* __restrict__ s, _Float16* __restrict__ outh,
    float* __restrict__ outf) {
  constexpr int KX     = CINX * TAPS;
  constexpr int KXP    = (KX + 31) / 32 * 32;
  constexpr int KS9    = CINS * TAPS;
  constexpr int KSTEPS = (KXP + KS9 + 31) / 32;
  constexpr int HX     = OH * STRIDE;
  constexpr int WX     = OW * STRIDE;
  constexpr int NPIX   = OH * OW;
  constexpr int BH     = 320 / OW;
  constexpr int PADL   = (TAPS == 9 && STRIDE == 1) ? 1 : 0;
  constexpr int ROWSX  = (TAPS == 9) ? (BH * STRIDE + (STRIDE == 1 ? 2 : 1)) : BH;
  constexpr int WPADX  = WX + 2;
  constexpr int LDSX   = CINX * ROWSX * WPADX;
  constexpr int ROWSS  = BH + 2;
  constexpr int WPADS  = OW + 2;
  constexpr int LDSS   = (CINS > 0) ? CINS * ROWSS * WPADS : 0;
  constexpr int LGX    = (CINX == 8) ? 3 : (CINX == 16) ? 4 : (CINX == 32) ? 5 : 6;
  constexpr int LGS    = (CINS == 8) ? 3 : (CINS == 16) ? 4 : (CINS == 32) ? 5 : 6;

  extern __shared__ _Float16 smem[];
  _Float16* ldsx = smem;
  _Float16* ldss = smem + LDSX;

  int mtile = blockIdx.y;
  int oy0   = blockIdx.x * BH;

  // ---- stage x band into LDS (halo pre-zeroed) ----
  {
    const int iy0 = oy0 * STRIDE - PADL;
    for (int i = threadIdx.x; i < LDSX; i += 160) {
      int c   = i / (ROWSX * WPADX);
      int rem = i - c * (ROWSX * WPADX);
      int r   = rem / WPADX;
      int co  = rem - r * WPADX;
      int iy  = iy0 + r;
      int gx  = co - PADL;
      _Float16 v = (_Float16)0.f;
      if (iy >= 0 && iy < HX && gx >= 0 && gx < WX)
        v = x[((size_t)c * HX + iy) * WX + gx];
      ldsx[(r * WPADX + co) * CINX + c] = v;
    }
  }
  // ---- stage s band into LDS (stride 1, pad 1) ----
  if constexpr (CINS > 0) {
    const int iy0s = oy0 - 1;
    for (int i = threadIdx.x; i < LDSS; i += 160) {
      int c   = i / (ROWSS * WPADS);
      int rem = i - c * (ROWSS * WPADS);
      int r   = rem / WPADS;
      int co  = rem - r * WPADS;
      int iy  = iy0s + r;
      int gx  = co - 1;
      _Float16 v = (_Float16)0.f;
      if (iy >= 0 && iy < OH && gx >= 0 && gx < OW)
        v = s[((size_t)c * OH + iy) * OW + gx];
      ldss[(r * WPADS + co) * CINS + c] = v;
    }
  }
  __syncthreads();

  int lane = threadIdx.x & 31;
  int wv   = threadIdx.x >> 5;
  int nloc = lane & 15, hi = lane >> 4;

  int oyl[4], oxv[4];
#pragma unroll
  for (int t = 0; t < 4; ++t) {
    int lp = (wv * 4 + t) * 16 + nloc;   // 0..319 within band
    oyl[t] = lp / OW;
    oxv[t] = lp - oyl[t] * OW;
  }

  const _Float16* abase = Apk + ((size_t)(mtile * KSTEPS) * 32 + lane) * 16;

  v8f acc[4];
#pragma unroll
  for (int t = 0; t < 4; ++t) acc[t] = (v8f){};

  for (int ksi = 0; ksi < KSTEPS; ++ksi) {
    v16h a = *(const v16h*)(abase + (size_t)ksi * 512);
    if (ksi + 1 < KSTEPS)
      __builtin_prefetch(abase + (size_t)(ksi + 1) * 512, 0, 0);
#pragma unroll
    for (int t = 0; t < 4; ++t) {
      union { v16h v; v8h g[2]; } b;
#pragma unroll
      for (int g = 0; g < 2; ++g) {
        int kb = ksi * 32 + g * 16 + hi * 8;   // 8-aligned run: one tap, one region
        v8h f = (v8h){};
        if (kb < KX) {
          int tap = kb >> LGX;
          int c0  = kb & (CINX - 1);
          int dy, dx;
          if constexpr (TAPS == 9) { dy = tap / 3; dx = tap - dy * 3; }
          else                     { dy = 0; dx = 0; }
          int ly = oyl[t] * STRIDE + dy;
          int lx = oxv[t] * STRIDE + dx;
          f = *(const v8h*)(ldsx + ((ly * WPADX + lx) * CINX + c0));
        } else if constexpr (CINS > 0) {
          if (kb >= KXP) {
            int ks = kb - KXP;
            if (ks < KS9) {
              int tap = ks >> LGS;
              int c0  = ks & (CINS - 1);
              int dy = tap / 3, dx = tap - dy * 3;
              int ly = oyl[t] + dy;
              int lx = oxv[t] + dx;
              f = *(const v8h*)(ldss + ((ly * WPADS + lx) * CINS + c0));
            }
          }
        }
        b.g[g] = f;
      }
      acc[t] = __builtin_amdgcn_wmma_f32_16x16x32_f16(false, a, false, b.v,
                                                      (short)0, acc[t], false, false);
    }
  }

#pragma unroll
  for (int t = 0; t < 4; ++t) {
    int p = blockIdx.x * 320 + (wv * 4 + t) * 16 + nloc;   // global pixel
#pragma unroll
    for (int r = 0; r < 8; ++r) {
      int ch = mtile * 16 + r + 8 * hi;   // D layout: VGPR r -> M = r + 8*hi
      if (ch < COUT) {
        float v = acc[t][r];
        if constexpr (RELU) v = fmaxf(v, 0.f);
        if constexpr (OUTH_EN) outh[(size_t)ch * NPIX + p] = (_Float16)v;
        if constexpr (OUTF_EN) outf[(size_t)ch * NPIX + p] = v;
      }
    }
  }
}

// ---------------------------------------------------------------------------
// u = cprev + up2(t) ; cprev,u at (C,Ho,Wo) f16, t at (C,Ho/2,Wo/2) f32
// ---------------------------------------------------------------------------
__global__ void upsample_add_kernel(const _Float16* __restrict__ cprev,
                                    const float* __restrict__ t,
                                    _Float16* __restrict__ u, int C, int Ho, int Wo) {
  int i = blockIdx.x * blockDim.x + threadIdx.x;
  int tot = C * Ho * Wo;
  if (i >= tot) return;
  int xx = i % Wo;
  int yy = (i / Wo) % Ho;
  int c  = i / (Wo * Ho);
  int Hh = Ho >> 1, Wh = Wo >> 1;
  float v = (float)cprev[i] + t[((size_t)c * Hh + (yy >> 1)) * Wh + (xx >> 1)];
  u[i] = (_Float16)v;
}

// ---------------------------------------------------------------------------
// Softmax-over-depth accumulation
// ---------------------------------------------------------------------------
__global__ void accum_kernel(const float* __restrict__ cost,
                             const float* __restrict__ dvals, int d,
                             float* __restrict__ exp_sum, float* __restrict__ depth_img,
                             float* __restrict__ max_prob) {
  int p = blockIdx.x * blockDim.x + threadIdx.x;
  if (p >= N1) return;
  float prob = expf(cost[p]);
  exp_sum[p]  += prob;
  depth_img[p] += dvals[d * N1 + p] * prob;
  max_prob[p]  = fmaxf(max_prob[p], prob);
}

__global__ void finalize_kernel(const float* __restrict__ exp_sum,
                                const float* __restrict__ depth_img,
                                const float* __restrict__ max_prob,
                                float* __restrict__ out) {
  int p = blockIdx.x * blockDim.x + threadIdx.x;
  if (p >= N1) return;
  float es = exp_sum[p] + 1e-10f;
  out[p]      = depth_img[p] / es;
  out[N1 + p] = max_prob[p] / es;
}

// ---------------------------------------------------------------------------
// Host helpers mirroring kernel constexprs
// ---------------------------------------------------------------------------
static int ksteps_of(int taps, int cinx, int cins) {
  int kx  = taps * cinx;
  int kxp = (kx + 31) / 32 * 32;
  return (kxp + taps * cins + 31) / 32;
}
static int lds_halves(int taps, int stride, int cinx, int cins, int ow) {
  int bh    = 320 / ow;
  int rowsx = (taps == 9) ? (bh * stride + (stride == 1 ? 2 : 1)) : bh;
  int wpadx = ow * stride + 2;
  int tot   = cinx * rowsx * wpadx;
  if (cins > 0) tot += cins * (bh + 2) * (ow + 2);
  return tot;
}
static int lg2i(int v) { return v == 8 ? 3 : v == 16 ? 4 : v == 32 ? 5 : 6; }

// ---------------------------------------------------------------------------
// Host orchestration
// ---------------------------------------------------------------------------
extern "C" void kernel_launch(void* const* d_in, const int* in_sizes, int n_in,
                              void* d_out, int out_size, void* d_ws, size_t ws_size,
                              hipStream_t stream) {
  (void)in_sizes; (void)n_in; (void)out_size; (void)ws_size;
  const float* feats = (const float*)d_in[0];
  const float* proj  = (const float*)d_in[1];
  const float* dvals = (const float*)d_in[2];
  const float* W0   = (const float*)d_in[3];
  const float* Ws1  = (const float*)d_in[4];
  const float* W1   = (const float*)d_in[5];
  const float* Ws2  = (const float*)d_in[6];
  const float* W2   = (const float*)d_in[7];
  const float* Ws3  = (const float*)d_in[8];
  const float* W3   = (const float*)d_in[9];
  const float* Ws4  = (const float*)d_in[10];
  const float* U4   = (const float*)d_in[11];
  const float* U3   = (const float*)d_in[12];
  const float* U2   = (const float*)d_in[13];
  const float* Wout = (const float*)d_in[14];
  float* outp = (float*)d_out;

  const int n1 = N1;
  const int n2 = 160 * 160;
  const int n3 = 80 * 80;
  const int n4 = 40 * 40;

  const int KS_L1 = ksteps_of(9, 32, 8);    // 12
  const int KS_L2 = ksteps_of(9, 8, 16);    // 8
  const int KS_L3 = ksteps_of(9, 16, 32);   // 14
  const int KS_L4 = ksteps_of(9, 32, 64);   // 27
  const int KS_U4 = ksteps_of(1, 64, 0);    // 2
  const int KS_U3 = ksteps_of(1, 32, 0);    // 1
  const int KS_U2 = ksteps_of(1, 16, 0);    // 1
  const int KS_WO = ksteps_of(9, 8, 0);     // 3

  uint8_t* base = (uint8_t*)d_ws;
  size_t off = 0;
  auto alloc = [&](size_t bytes) -> size_t {
    size_t o = off;
    off = (off + bytes + 255) & ~(size_t)255;
    return o;
  };
  size_t o_rt   = alloc(24 * 4);
  size_t o_aL1  = alloc((size_t)1 * KS_L1 * 512 * 2);
  size_t o_aL2  = alloc((size_t)1 * KS_L2 * 512 * 2);
  size_t o_aL3  = alloc((size_t)2 * KS_L3 * 512 * 2);
  size_t o_aL4  = alloc((size_t)4 * KS_L4 * 512 * 2);
  size_t o_aU4  = alloc((size_t)2 * KS_U4 * 512 * 2);
  size_t o_aU3  = alloc((size_t)1 * KS_U3 * 512 * 2);
  size_t o_aU2  = alloc((size_t)1 * KS_U2 * 512 * 2);
  size_t o_aWO  = alloc((size_t)1 * KS_WO * 512 * 2);
  size_t o_var  = alloc((size_t)32 * n1 * 2);
  size_t o_s1a  = alloc((size_t)8  * n1 * 2);
  size_t o_s2a  = alloc((size_t)16 * n2 * 2);
  size_t o_s3a  = alloc((size_t)32 * n3 * 2);
  size_t o_s4a  = alloc((size_t)64 * n4 * 2);
  size_t o_s1b  = alloc((size_t)8  * n1 * 2);
  size_t o_s2b  = alloc((size_t)16 * n2 * 2);
  size_t o_s3b  = alloc((size_t)32 * n3 * 2);
  size_t o_s4b  = alloc((size_t)64 * n4 * 2);
  size_t o_u3   = alloc((size_t)32 * n3 * 2);
  size_t o_u2   = alloc((size_t)16 * n2 * 2);
  size_t o_u1   = alloc((size_t)8  * n1 * 2);
  size_t o_t4   = alloc((size_t)32 * n4 * 4);
  size_t o_t3   = alloc((size_t)16 * n3 * 4);
  size_t o_t2   = alloc((size_t)8  * n2 * 4);
  size_t o_cost = alloc((size_t)n1 * 4);
  size_t o_esum = alloc((size_t)n1 * 4);
  size_t o_dimg = alloc((size_t)n1 * 4);
  size_t o_mprb = alloc((size_t)n1 * 4);

  float*    rt   = (float*)(base + o_rt);
  _Float16* aL1  = (_Float16*)(base + o_aL1);
  _Float16* aL2  = (_Float16*)(base + o_aL2);
  _Float16* aL3  = (_Float16*)(base + o_aL3);
  _Float16* aL4  = (_Float16*)(base + o_aL4);
  _Float16* aU4  = (_Float16*)(base + o_aU4);
  _Float16* aU3  = (_Float16*)(base + o_aU3);
  _Float16* aU2  = (_Float16*)(base + o_aU2);
  _Float16* aWO  = (_Float16*)(base + o_aWO);
  _Float16* var  = (_Float16*)(base + o_var);
  _Float16* s1   = (_Float16*)(base + o_s1a);
  _Float16* s2   = (_Float16*)(base + o_s2a);
  _Float16* s3   = (_Float16*)(base + o_s3a);
  _Float16* s4   = (_Float16*)(base + o_s4a);
  _Float16* c1   = (_Float16*)(base + o_s1b);
  _Float16* c2   = (_Float16*)(base + o_s2b);
  _Float16* c3   = (_Float16*)(base + o_s3b);
  _Float16* c4   = (_Float16*)(base + o_s4b);
  _Float16* u3b  = (_Float16*)(base + o_u3);
  _Float16* u2b  = (_Float16*)(base + o_u2);
  _Float16* u1b  = (_Float16*)(base + o_u1);
  float*    t4   = (float*)(base + o_t4);
  float*    t3   = (float*)(base + o_t3);
  float*    t2   = (float*)(base + o_t2);
  float*    cost = (float*)(base + o_cost);
  float*    esum = (float*)(base + o_esum);
  float*    dimg = (float*)(base + o_dimg);
  float*    mprb = (float*)(base + o_mprb);

  auto zero = [&](void* p, size_t bytes) {
    int n = (int)(bytes / 4);
    zero_kernel<<<(n + 255) / 256, 256, 0, stream>>>((uint32_t*)p, n);
  };

  zero(s1, (size_t)8  * n1 * 2);
  zero(s2, (size_t)16 * n2 * 2);
  zero(s3, (size_t)32 * n3 * 2);
  zero(s4, (size_t)64 * n4 * 2);
  zero(esum, (size_t)n1 * 4);
  zero(dimg, (size_t)n1 * 4);
  zero(mprb, (size_t)n1 * 4);

  prep_proj_kernel<<<1, 32, 0, stream>>>(proj, rt);

  auto pack = [&](const float* Wx, const float* Ws, int Cout, int Cinx, int Cins,
                  int taps, int ksteps, int mtiles, _Float16* dst) {
    int kx  = taps * Cinx;
    int kxp = (kx + 31) / 32 * 32;
    int total = mtiles * ksteps * 512;
    pack_weights_kernel<<<(total + 255) / 256, 256, 0, stream>>>(
        Wx, Ws, Cout, Cinx, lg2i(Cinx), Cins, Cins > 0 ? lg2i(Cins) : 0,
        taps, kx, kxp, ksteps, mtiles, dst);
  };
  pack(W0,   Ws1, 8,  32, 8,  9, KS_L1, 1, aL1);
  pack(W1,   Ws2, 16, 8,  16, 9, KS_L2, 1, aL2);
  pack(W2,   Ws3, 32, 16, 32, 9, KS_L3, 2, aL3);
  pack(W3,   Ws4, 64, 32, 64, 9, KS_L4, 4, aL4);
  pack(U4,   nullptr, 32, 64, 0, 1, KS_U4, 2, aU4);
  pack(U3,   nullptr, 16, 32, 0, 1, KS_U3, 1, aU3);
  pack(U2,   nullptr, 8,  16, 0, 1, KS_U2, 1, aU2);
  pack(Wout, nullptr, 1,  8,  0, 9, KS_WO, 1, aWO);

  const size_t sh_L1 = (size_t)lds_halves(9, 1, 32, 8,  320) * 2;
  const size_t sh_L2 = (size_t)lds_halves(9, 2, 8,  16, 160) * 2;
  const size_t sh_L3 = (size_t)lds_halves(9, 2, 16, 32, 80)  * 2;
  const size_t sh_L4 = (size_t)lds_halves(9, 2, 32, 64, 40)  * 2;
  const size_t sh_U4 = (size_t)lds_halves(1, 1, 64, 0,  40)  * 2;
  const size_t sh_U3 = (size_t)lds_halves(1, 1, 32, 0,  80)  * 2;
  const size_t sh_U2 = (size_t)lds_halves(1, 1, 16, 0,  160) * 2;
  const size_t sh_WO = (size_t)lds_halves(9, 1, 8,  0,  320) * 2;

  for (int d = 0; d < 32; ++d) {
    warp_variance_kernel<<<(n1 + 255) / 256, 256, 0, stream>>>(feats, rt, dvals, d, var);

    conv_wmma_kernel<9, 1, 32, 8, 8, 320, 320, true, true, false>
        <<<dim3(320, 1), 160, sh_L1, stream>>>(aL1, var, s1, c1, nullptr);
    conv_wmma_kernel<9, 2, 8, 16, 16, 160, 160, true, true, false>
        <<<dim3(80, 1), 160, sh_L2, stream>>>(aL2, c1, s2, c2, nullptr);
    conv_wmma_kernel<9, 2, 16, 32, 32, 80, 80, true, true, false>
        <<<dim3(20, 2), 160, sh_L3, stream>>>(aL3, c2, s3, c3, nullptr);
    conv_wmma_kernel<9, 2, 32, 64, 64, 40, 40, true, true, false>
        <<<dim3(5, 4), 160, sh_L4, stream>>>(aL4, c3, s4, c4, nullptr);

    conv_wmma_kernel<1, 1, 64, 0, 32, 40, 40, false, false, true>
        <<<dim3(5, 2), 160, sh_U4, stream>>>(aU4, c4, nullptr, nullptr, t4);
    upsample_add_kernel<<<(32 * n3 + 255) / 256, 256, 0, stream>>>(c3, t4, u3b, 32, 80, 80);
    conv_wmma_kernel<1, 1, 32, 0, 16, 80, 80, false, false, true>
        <<<dim3(20, 1), 160, sh_U3, stream>>>(aU3, u3b, nullptr, nullptr, t3);
    upsample_add_kernel<<<(16 * n2 + 255) / 256, 256, 0, stream>>>(c2, t3, u2b, 16, 160, 160);
    conv_wmma_kernel<1, 1, 16, 0, 8, 160, 160, false, false, true>
        <<<dim3(80, 1), 160, sh_U2, stream>>>(aU2, u2b, nullptr, nullptr, t2);
    upsample_add_kernel<<<(8 * n1 + 255) / 256, 256, 0, stream>>>(c1, t2, u1b, 8, 320, 320);
    conv_wmma_kernel<9, 1, 8, 0, 1, 320, 320, false, false, true>
        <<<dim3(320, 1), 160, sh_WO, stream>>>(aWO, u1b, nullptr, nullptr, cost);

    accum_kernel<<<(n1 + 255) / 256, 256, 0, stream>>>(cost, dvals, d, esum, dimg, mprb);

    _Float16* t;
    t = s1; s1 = c1; c1 = t;
    t = s2; s2 = c2; c2 = t;
    t = s3; s3 = c3; c3 = t;
    t = s4; s4 = c4; c4 = t;
  }

  finalize_kernel<<<(n1 + 255) / 256, 256, 0, stream>>>(esum, dimg, mprb, outp);
}
